// MultiHeadSelfAttention_71451075936287
// MI455X (gfx1250) — matrix operand order repaired and checked by reading, not compiled, not run on hardware
//
#include <hip/hip_runtime.h>

// MI455X / gfx1250, wave32. bf16 WMMA + async global->LDS staging (double buffered).
// Workspace layout (needs ~92 MB):
//   xb     @ 0         : B*S*E bf16        (16 MB)
//   wqkvT  @ 16777216  : 3E x E bf16       ( 6 MB)
//   wprojT @ 23068672  : E x E bf16        ( 2 MB)
//   Q      @ 25165824  : (b,h,s,d) bf16    (16 MB)
//   K      @ 41943040  : (b,h,s,d) bf16    (16 MB)
//   Vt     @ 58720256  : (b,h,d,s) bf16    (16 MB)
//   attn   @ 75497472  : (b,s,e) bf16      (16 MB)

typedef __attribute__((ext_vector_type(16))) __bf16 v16bf;
typedef __attribute__((ext_vector_type(8)))  __bf16 v8bf;
typedef __attribute__((ext_vector_type(4)))  __bf16 v4bf;
typedef __attribute__((ext_vector_type(8)))  float  v8f;

#define EMBED 1024
#define SEQ   2048
#define BATCH 4
#define HEADS 16
#define HDIM  64
#define NQKV  3072

static __device__ __forceinline__ v16bf mk16(v8bf lo, v8bf hi) {
  return __builtin_shufflevector(lo, hi, 0,1,2,3,4,5,6,7,8,9,10,11,12,13,14,15);
}

// CDNA5 async DMA: global -> LDS, 16 bytes per lane, tracked by ASYNCcnt.
static __device__ __forceinline__ void async_b128(unsigned lds_off,
                                                  unsigned long long gaddr) {
  asm volatile("global_load_async_to_lds_b128 %0, %1, off"
               :: "v"(lds_off), "v"(gaddr) : "memory");
}
#define WAIT_ASYNC(n) asm volatile("s_wait_asynccnt %0" :: "i"(n) : "memory")

static __device__ __forceinline__ unsigned lds_off32(const void* p) {
  return (unsigned)(size_t)p;   // AS3 offset lives in the low 32 bits
}

// ---------------- fp32 -> bf16 copy ----------------
__global__ __launch_bounds__(256) void cvt_bf16(const float* __restrict__ in,
                                                __bf16* __restrict__ out) {
  size_t i = ((size_t)blockIdx.x * 256 + threadIdx.x) * 4;
  float4 f = *(const float4*)(in + i);
  v4bf r = { (__bf16)f.x, (__bf16)f.y, (__bf16)f.z, (__bf16)f.w };
  *(v4bf*)(out + i) = r;
}

// ------------- fp32 RxC -> bf16 CxR transpose -------------
__global__ __launch_bounds__(256) void transpose_cvt(const float* __restrict__ W,
                                                     __bf16* __restrict__ Wt,
                                                     int R, int C) {
  __shared__ float t[16][17];
  int c0 = blockIdx.x * 16, r0 = blockIdx.y * 16;
  int tx = threadIdx.x, ty = threadIdx.y;
  t[ty][tx] = W[(size_t)(r0 + ty) * C + c0 + tx];
  __syncthreads();
  Wt[(size_t)(c0 + ty) * R + r0 + tx] = (__bf16)t[tx][ty];
}

// ------------- bf16 WMMA GEMM:  C[M,N] = A[M,K] @ Bt[N,K]^T + bias -------------
// 128x128 block tile, 8 waves of 32x64, async double-buffered LDS staging.
// mode 0: fp32 row-major output to Cf (projection)
// mode 1: QKV scatter: col<1024 -> Q(b,h,s,d); <2048 -> K(b,h,s,d); else V^T(b,h,d,s)
#define BM 128
#define BN 128
#define BK 32
__global__ __launch_bounds__(256) void gemm_bf16_wmma(
    const __bf16* __restrict__ A, const __bf16* __restrict__ Bt,
    const float* __restrict__ bias, int Ndim, int Kdim, int mode,
    float* __restrict__ Cf, __bf16* __restrict__ Qo,
    __bf16* __restrict__ Ko, __bf16* __restrict__ Vto) {
  __shared__ __align__(16) __bf16 sA[2][BM * BK];   // 2 x 8 KB
  __shared__ __align__(16) __bf16 sB[2][BN * BK];   // 2 x 8 KB
  const int tid  = threadIdx.x;
  const int lane = tid & 31, wid = tid >> 5;
  const int lm   = lane & 15, hi = lane >> 4;
  const int n0   = blockIdx.x * BN;
  const int m0   = blockIdx.y * BM;
  const int wm   = wid & 3, wn = wid >> 2;

  // each thread issues 4 async b128: 2 for the A tile, 2 for the B tile
  auto stage = [&](int buf, int k0) {
#pragma unroll
    for (int p = 0; p < 2; ++p) {
      int c = tid + p * 256;                 // A: 512 chunks of 8 halves
      int r = c >> 2, kc = (c & 3) * 8;
      async_b128(lds_off32(sA[buf] + r * BK + kc),
                 (unsigned long long)(size_t)(A + (size_t)(m0 + r) * Kdim + k0 + kc));
    }
#pragma unroll
    for (int p = 0; p < 2; ++p) {
      int c = tid + p * 256;                 // B: 512 chunks
      int r = c >> 2, kc = (c & 3) * 8;
      async_b128(lds_off32(sB[buf] + r * BK + kc),
                 (unsigned long long)(size_t)(Bt + (size_t)(n0 + r) * Kdim + k0 + kc));
    }
  };

  v8f acc[2][4] = {};
  const int ktiles = Kdim / BK;
  stage(0, 0);
  for (int i = 0; i < ktiles; ++i) {
    int k0n = (i + 1 < ktiles) ? (i + 1) * BK : 0;   // wrap: keeps counted wait exact
    stage((i + 1) & 1, k0n);
    WAIT_ASYNC(4);                                    // group i done, group i+1 in flight
    __syncthreads();
    const __bf16* a = sA[i & 1];
    const __bf16* b = sB[i & 1];
    v16bf af[2], bfr[4];
#pragma unroll
    for (int tm = 0; tm < 2; ++tm) {       // A frag: 16x32 row-major
      const __bf16* p = a + (wm * 32 + tm * 16 + lm) * BK + hi * 8;
      af[tm] = mk16(*(const v8bf*)p, *(const v8bf*)(p + 16));
    }
#pragma unroll
    for (int tn = 0; tn < 4; ++tn) {       // B frag: 32x16 from N-major tile
      const __bf16* p = b + (wn * 64 + tn * 16 + lm) * BK + hi * 8;
      bfr[tn] = mk16(*(const v8bf*)p, *(const v8bf*)(p + 16));
    }
#pragma unroll
    for (int tm = 0; tm < 2; ++tm)
#pragma unroll
      for (int tn = 0; tn < 4; ++tn)
        acc[tm][tn] = __builtin_amdgcn_wmma_f32_16x16x32_bf16(
            false, af[tm], false, bfr[tn], (short)0, acc[tm][tn], false, false);
    __syncthreads();                        // reads done before buffer reuse
  }

  // epilogue: C layout row M = v + 8*hi, col N = lane%16
#pragma unroll
  for (int tm = 0; tm < 2; ++tm) {
#pragma unroll
    for (int tn = 0; tn < 4; ++tn) {
      int col = n0 + wn * 64 + tn * 16 + lm;
      float bv = bias[col];
#pragma unroll
      for (int v = 0; v < 8; ++v) {
        int row = m0 + wm * 32 + tm * 16 + v + 8 * hi;
        float val = acc[tm][tn][v] + bv;
        if (mode == 0) {
          Cf[(size_t)row * Ndim + col] = val;
        } else {
          int sec = col >> 10, cc = col & 1023;
          int h = cc >> 6, d = cc & 63;
          int b = row >> 11, s = row & 2047;
          int bh = b * HEADS + h;
          if (sec == 0)      Qo[((size_t)bh * SEQ + s) * HDIM + d]  = (__bf16)val;
          else if (sec == 1) Ko[((size_t)bh * SEQ + s) * HDIM + d]  = (__bf16)val;
          else               Vto[((size_t)bh * HDIM + d) * SEQ + s] = (__bf16)val;
        }
      }
    }
  }
}

// ------------- flash attention -------------
// 4 waves x 16 query rows; K/V 64-blocks async-staged to LDS once per workgroup
// (shared by all 4 waves), double buffered with depth-1 prefetch.
__global__ __launch_bounds__(128) void attn_wmma(
    const __bf16* __restrict__ Q, const __bf16* __restrict__ K,
    const __bf16* __restrict__ Vt, __bf16* __restrict__ Out) {
  __shared__ __align__(16) __bf16 sK[2][64 * 64];   // 2 x 8 KB (key, d)
  __shared__ __align__(16) __bf16 sV[2][64 * 64];   // 2 x 8 KB (d, key)
  __shared__ __align__(16) __bf16 sP[4][16 * 64];   // per-wave P tile
  const int tid = threadIdx.x, lane = tid & 31, wid = tid >> 5;
  const int lm = lane & 15, hi = lane >> 4;
  const int bh = blockIdx.y;
  const int q0 = (blockIdx.x * 4 + wid) * 16;
  const float SC = 0.125f * 1.44269504088896f;      // 1/sqrt(64) * log2(e)

  const __bf16* qb = Q  + (size_t)bh * SEQ * HDIM;
  const __bf16* kp = K  + (size_t)bh * SEQ * HDIM;
  const __bf16* vp = Vt + (size_t)bh * HDIM * SEQ;
  __bf16* pbuf = sP[wid];

  // each thread: 8 async b128 per key block (4 for K tile, 4 for V tile)
  auto stage = [&](int buf, int kb) {
#pragma unroll
    for (int p = 0; p < 4; ++p) {
      int c = tid + p * 128;                 // 512 chunks per tile
      int r = c >> 3, o = (c & 7) * 8;
      async_b128(lds_off32(sK[buf] + r * 64 + o),
                 (unsigned long long)(size_t)(kp + (size_t)(kb + r) * HDIM + o));
      async_b128(lds_off32(sV[buf] + r * 64 + o),
                 (unsigned long long)(size_t)(vp + (size_t)r * SEQ + kb + o));
    }
  };

  // Q fragments held in registers for the whole pass (16x64 = 2 A-frags)
  v16bf aq[2];
  {
    const __bf16* p = qb + (size_t)(q0 + lm) * HDIM;
#pragma unroll
    for (int t = 0; t < 2; ++t) {
      const __bf16* pp = p + t * 32 + hi * 8;
      aq[t] = mk16(*(const v8bf*)pp, *(const v8bf*)(pp + 16));
    }
  }

  float m[8], l[8];
  v8f o[4] = {};
#pragma unroll
  for (int v = 0; v < 8; ++v) { m[v] = -1e30f; l[v] = 0.f; }

  stage(0, 0);
  for (int i = 0; i < SEQ / 64; ++i) {
    const int kb0 = i * 64;
    stage((i + 1) & 1, (kb0 + 64) & (SEQ - 1));      // wrap keeps counted wait exact
    WAIT_ASYNC(8);
    __syncthreads();
    const __bf16* kt = sK[i & 1];
    const __bf16* vt = sV[i & 1];

    // S = Q @ K^T : K tile rows are N-major B fragments
    v8f sc[4];
#pragma unroll
    for (int nt = 0; nt < 4; ++nt) {
      const __bf16* p = kt + (nt * 16 + lm) * 64;
      v8f c = {};
#pragma unroll
      for (int t = 0; t < 2; ++t) {
        const __bf16* pp = p + t * 32 + hi * 8;
        v16bf bk = mk16(*(const v8bf*)pp, *(const v8bf*)(pp + 16));
        c = __builtin_amdgcn_wmma_f32_16x16x32_bf16(false, aq[t], false, bk,
                                                    (short)0, c, false, false);
      }
      sc[nt] = c;
    }

    // online softmax: row v+8*hi lives in one 16-lane half; butterfly in-half
#pragma unroll
    for (int v = 0; v < 8; ++v) {
#pragma unroll
      for (int nt = 0; nt < 4; ++nt) sc[nt][v] *= SC;
      float t0 = fmaxf(fmaxf(sc[0][v], sc[1][v]), fmaxf(sc[2][v], sc[3][v]));
#pragma unroll
      for (int off = 8; off >= 1; off >>= 1)
        t0 = fmaxf(t0, __shfl_xor(t0, off, 32));
      float mn   = fmaxf(m[v], t0);
      float corr = exp2f(m[v] - mn);
      float rs   = 0.f;
#pragma unroll
      for (int nt = 0; nt < 4; ++nt) {
        float pv = exp2f(sc[nt][v] - mn);
        sc[nt][v] = pv;
        rs += pv;
      }
#pragma unroll
      for (int off = 8; off >= 1; off >>= 1) rs += __shfl_xor(rs, off, 32);
      m[v] = mn;
      l[v] = l[v] * corr + rs;
#pragma unroll
      for (int j = 0; j < 4; ++j) o[j][v] *= corr;
    }

    // C-layout P -> LDS (row-major 16x64 bf16) -> A-fragment layout
#pragma unroll
    for (int nt = 0; nt < 4; ++nt)
#pragma unroll
      for (int v = 0; v < 8; ++v)
        pbuf[(v + 8 * hi) * 64 + nt * 16 + lm] = (__bf16)sc[nt][v];

    v16bf pa[2];
#pragma unroll
    for (int t = 0; t < 2; ++t) {
      const __bf16* pp = pbuf + lm * 64 + t * 32 + hi * 8;
      pa[t] = mk16(*(const v8bf*)pp, *(const v8bf*)(pp + 16));
    }

    // O += P @ V : V tile rows (d-major) are N-major B fragments
#pragma unroll
    for (int j = 0; j < 4; ++j) {
      const __bf16* vr = vt + (j * 16 + lm) * 64;
#pragma unroll
      for (int t = 0; t < 2; ++t) {
        const __bf16* pp = vr + t * 32 + hi * 8;
        v16bf bv = mk16(*(const v8bf*)pp, *(const v8bf*)(pp + 16));
        o[j] = __builtin_amdgcn_wmma_f32_16x16x32_bf16(false, pa[t], false, bv,
                                                       (short)0, o[j], false, false);
      }
    }
    __syncthreads();          // everyone done reading buf before it is re-staged
  }

  // epilogue: normalize and store to (b, s, e) bf16
  const int b = bh >> 4, h = bh & 15;
#pragma unroll
  for (int j = 0; j < 4; ++j)
#pragma unroll
    for (int v = 0; v < 8; ++v) {
      int row = q0 + v + 8 * hi;
      float val = o[j][v] / l[v];
      Out[((size_t)(b * SEQ + row)) * EMBED + h * HDIM + j * 16 + lm] = (__bf16)val;
    }
}

extern "C" void kernel_launch(void* const* d_in, const int* in_sizes, int n_in,
                              void* d_out, int out_size, void* d_ws, size_t ws_size,
                              hipStream_t stream) {
  const float* x     = (const float*)d_in[0];
  const float* Wqkv  = (const float*)d_in[1];
  const float* bqkv  = (const float*)d_in[2];
  const float* Wproj = (const float*)d_in[3];
  const float* bproj = (const float*)d_in[4];
  float* out = (float*)d_out;
  char* ws = (char*)d_ws;

  __bf16* xb     = (__bf16*)(ws);
  __bf16* wqkvT  = (__bf16*)(ws + 16777216);
  __bf16* wprojT = (__bf16*)(ws + 23068672);
  __bf16* q      = (__bf16*)(ws + 25165824);
  __bf16* k      = (__bf16*)(ws + 41943040);
  __bf16* vt     = (__bf16*)(ws + 58720256);
  __bf16* attnb  = (__bf16*)(ws + 75497472);

  // 1) x -> bf16
  cvt_bf16<<<8192, 256, 0, stream>>>(x, xb);
  // 2) weights -> transposed bf16 (N x K)
  transpose_cvt<<<dim3(NQKV / 16, EMBED / 16), dim3(16, 16), 0, stream>>>(
      Wqkv, wqkvT, EMBED, NQKV);
  transpose_cvt<<<dim3(EMBED / 16, EMBED / 16), dim3(16, 16), 0, stream>>>(
      Wproj, wprojT, EMBED, EMBED);
  // 3) QKV GEMM with scatter epilogue (V transposed)
  gemm_bf16_wmma<<<dim3(NQKV / BN, BATCH * SEQ / BM), 256, 0, stream>>>(
      xb, wqkvT, bqkv, NQKV, EMBED, 1, nullptr, q, k, vt);
  // 4) flash attention
  attn_wmma<<<dim3(SEQ / 64, BATCH * HEADS), 128, 0, stream>>>(q, k, vt, attnb);
  // 5) output projection -> fp32 d_out
  gemm_bf16_wmma<<<dim3(EMBED / BN, BATCH * SEQ / BM), 256, 0, stream>>>(
      attnb, wprojT, bproj, EMBED, EMBED, 0, out, nullptr, nullptr, nullptr);
}
